// FeatureAttention_53944789238186
// MI455X (gfx1250) — matrix operand order repaired and checked by reading, not compiled, not run on hardware
//
#include <hip/hip_runtime.h>

// ---------------------------------------------------------------------------
// FeatureAttention for MI455X (gfx1250, wave32, WMMA).
//   Stage 1: QKV projection, f32 -> bf16, WMMA f32_16x16x32_bf16, V stored
//            transposed [B][E][S] for contiguous B-operand loads later.
//   Stage 2: fused flash attention with double-buffered K/V^T tiles staged
//            via GLOBAL_LOAD_ASYNC_TO_LDS_B128 (ASYNCcnt), one barrier/step.
// ---------------------------------------------------------------------------

typedef __attribute__((ext_vector_type(16))) __bf16 bf16x16;
typedef __attribute__((ext_vector_type(8)))  float  f32x8;

union FragBF { bf16x16 v; uint4 u[2]; };

constexpr int BATCH = 64;
constexpr int SEQ   = 1024;
constexpr int EMB   = 256;   // = HIDDEN
constexpr float LOG2E = 1.44269504088896340736f;

// padded LDS strides (elements). 264*2B = 528B: 16B aligned, and
// 528/4 = 132 words -> 132*m mod 64 distinct for m in [0,16): conflict-free.
constexpr int LD264 = 264;
// 40*2B = 80B: 16B aligned, 20*m mod 64 distinct for m in [0,16).
constexpr int LD40  = 40;

#define WMMA_BF16(A, B, C) \
    __builtin_amdgcn_wmma_f32_16x16x32_bf16(false, (A), false, (B), (short)0, (C), false, false)

// Async DMA: global memory -> LDS, 16B per lane, tracked by ASYNCcnt.
__device__ __forceinline__ void async_copy_b128(unsigned lds_addr,
                                                unsigned long long gaddr) {
    asm volatile("global_load_async_to_lds_b128 %0, %1, off"
                 :: "v"(lds_addr), "v"(gaddr) : "memory");
}
__device__ __forceinline__ void wait_async0() {
    asm volatile("s_wait_asynccnt 0x0" ::: "memory");
}

// ---------------------------------------------------------------------------
// Kernel 1: out[:, cols] = x @ W[:, cols] + b   (one of q/k/v per blockIdx.y)
// grid = (BS/128, 3*4): y = mat*4 + 64-col group. 256 threads = 8 waves.
// Each wave: one 16-row tile x four 16-col tiles, K = 256 in 8 WMMA steps.
// ---------------------------------------------------------------------------
__global__ __launch_bounds__(256)
void qkv_proj_kernel(const float* __restrict__ x,
                     const float* __restrict__ Wq, const float* __restrict__ bq,
                     const float* __restrict__ Wk, const float* __restrict__ bk,
                     const float* __restrict__ Wv, const float* __restrict__ bv,
                     __bf16* __restrict__ qo, __bf16* __restrict__ ko,
                     __bf16* __restrict__ vto)
{
    __shared__ __attribute__((aligned(16))) __bf16 xs[128 * LD264]; // x tile, bf16
    __shared__ __attribute__((aligned(16))) __bf16 wt[64 * LD264];  // W^T chunk, bf16

    const int tid  = threadIdx.x;
    const int mat  = blockIdx.y >> 2;       // 0:q 1:k 2:v
    const int cg   = blockIdx.y & 3;        // 64-column group
    const int row0 = blockIdx.x * 128;      // global row of x (flattened B*S)

    const float* W  = (mat == 0) ? Wq : (mat == 1) ? Wk : Wv;
    const float* bs = (mat == 0) ? bq : (mat == 1) ? bk : bv;

    // ---- stage x tile: 128 x 256 f32 -> bf16 row-major (padded) ----
#pragma unroll
    for (int i = 0; i < 32; ++i) {
        int idx = i * 256 + tid;                 // float4 index
        int r   = idx >> 6;
        int c4  = (idx & 63) << 2;
        const float4 f = *(const float4*)(x + (size_t)(row0 + r) * EMB + c4);
        __bf16* dst = &xs[r * LD264 + c4];
        dst[0] = (__bf16)f.x; dst[1] = (__bf16)f.y;
        dst[2] = (__bf16)f.z; dst[3] = (__bf16)f.w;
    }
    // ---- stage W^T chunk: wt[n][k] = W[k][cg*64+n], 256 x 64 ----
#pragma unroll
    for (int i = 0; i < 16; ++i) {
        int idx = i * 256 + tid;                 // float4 index
        int k   = idx >> 4;
        int nn  = (idx & 15) << 2;
        const float4 f = *(const float4*)(W + (size_t)k * EMB + cg * 64 + nn);
        wt[(nn + 0) * LD264 + k] = (__bf16)f.x;
        wt[(nn + 1) * LD264 + k] = (__bf16)f.y;
        wt[(nn + 2) * LD264 + k] = (__bf16)f.z;
        wt[(nn + 3) * LD264 + k] = (__bf16)f.w;
    }
    __syncthreads();

    const int wave = tid >> 5;
    const int lane = tid & 31;
    const int l16  = lane & 15;
    const int half = lane >> 4;
    const int m0   = wave * 16;

    // Cache A fragments (16 rows x K=256) in VGPRs.
    // A layout: lane row = lane%16; e<8 -> K = kt*32 + 8*half + e;
    //           e>=8 -> K = kt*32 + 16 + 8*half + (e-8).
    FragBF afrag[8];
    {
        const char* ab = (const char*)&xs[(m0 + l16) * LD264];
#pragma unroll
        for (int kt = 0; kt < 8; ++kt) {
            afrag[kt].u[0] = *(const uint4*)(ab + kt * 64 + 16 * half);
            afrag[kt].u[1] = *(const uint4*)(ab + kt * 64 + 32 + 16 * half);
        }
    }

#pragma unroll
    for (int ct = 0; ct < 4; ++ct) {
        const int n0 = ct * 16;
        f32x8 acc = {};
        // B layout: lane col n = lane%16; same K order as A.
        const char* bb = (const char*)&wt[(n0 + l16) * LD264];
#pragma unroll
        for (int kt = 0; kt < 8; ++kt) {
            FragBF bfrag;
            bfrag.u[0] = *(const uint4*)(bb + kt * 64 + 16 * half);
            bfrag.u[1] = *(const uint4*)(bb + kt * 64 + 32 + 16 * half);
            acc = WMMA_BF16(afrag[kt].v, bfrag.v, acc);
        }
        const int ncol = cg * 64 + n0 + l16;
        const float bias = bs[ncol];
        if (mat < 2) {
            __bf16* o = (mat == 0) ? qo : ko;
#pragma unroll
            for (int r = 0; r < 8; ++r) {
                size_t row = (size_t)(row0 + m0 + r + 8 * half);
                o[row * EMB + ncol] = (__bf16)(acc[r] + bias);
            }
        } else {
            // V stored transposed: vto[b][e][s]
            const int b  = row0 >> 10;
            const int sb = (row0 & 1023) + m0;
#pragma unroll
            for (int r = 0; r < 8; ++r) {
                vto[((size_t)b * EMB + ncol) * SEQ + sb + r + 8 * half] =
                    (__bf16)(acc[r] + bias);
            }
        }
    }
}

// ---------------------------------------------------------------------------
// Kernel 2: fused flash attention.
// grid = (SEQ/128, BATCH), 256 threads = 8 waves; wave owns 16 query rows,
// full E=256 ctx accumulator (16 e-tiles x 8 VGPRs), kv step = 32.
// K/V^T tiles double-buffered, filled by async global->LDS DMA.
// ---------------------------------------------------------------------------
__global__ __launch_bounds__(256)
void flash_attn_kernel(const __bf16* __restrict__ q,
                       const __bf16* __restrict__ k,
                       const __bf16* __restrict__ vt,
                       float* __restrict__ out)
{
    __shared__ __attribute__((aligned(16))) __bf16 qs[128 * LD264];      // q tile
    __shared__ __attribute__((aligned(16))) __bf16 ks[2][32 * LD264];    // k tiles
    __shared__ __attribute__((aligned(16))) __bf16 vs[2][256 * LD40];    // v^T tiles
    __shared__ __attribute__((aligned(16))) __bf16 ps[8 * 16 * LD40];    // P scratch

    const int tid  = threadIdx.x;
    const int wave = tid >> 5;
    const int lane = tid & 31;
    const int l16  = lane & 15;
    const int half = lane >> 4;

    const int b  = blockIdx.y;
    const int s0 = blockIdx.x * 128;
    const size_t rowbase = (size_t)b * SEQ + s0;     // global q row base

    // Issue async DMA of one kv tile (32 keys) into buffer bufi.
    auto stage_kv_async = [&](int t0, int bufi) {
        // k tile: 32 rows x 256 bf16 (4 x b128 per thread)
#pragma unroll
        for (int i = 0; i < 4; ++i) {
            int idx = i * 256 + tid;
            int tt  = idx >> 5;
            int c   = idx & 31;
            unsigned l = (unsigned)(size_t)&ks[bufi][tt * LD264] + c * 16;
            async_copy_b128(l, (unsigned long long)(size_t)
                                   (k + ((size_t)b * SEQ + t0 + tt) * EMB + c * 8));
        }
        // v^T tile: 256 rows x 32 bf16 (4 x b128 per thread)
#pragma unroll
        for (int i = 0; i < 4; ++i) {
            int idx = i * 256 + tid;
            int e   = idx >> 2;
            int c   = idx & 3;
            unsigned l = (unsigned)(size_t)&vs[bufi][e * LD40] + c * 16;
            async_copy_b128(l, (unsigned long long)(size_t)
                                   (vt + ((size_t)b * EMB + e) * SEQ + t0 + c * 8));
        }
    };

    // ---- stage q tile once: 128 rows x 256 bf16 ----
#pragma unroll
    for (int i = 0; i < 16; ++i) {
        int idx = i * 256 + tid;                     // uint4 index
        int r   = idx >> 5;
        int c   = idx & 31;
        uint4 d = *((const uint4*)(q + (rowbase + r) * EMB) + c);
        *(uint4*)((char*)qs + r * (LD264 * 2) + c * 16) = d;
    }
    // prefetch first kv tile
    stage_kv_async(0, 0);

    float m_run[8], l_run[8];
    f32x8 acc[16];
#pragma unroll
    for (int r = 0; r < 8; ++r) { m_run[r] = -__builtin_inff(); l_run[r] = 0.f; }
#pragma unroll
    for (int et = 0; et < 16; ++et) acc[et] = f32x8{};

    for (int t0 = 0; t0 < SEQ; t0 += 32) {
        const int bufi = (t0 >> 5) & 1;
        wait_async0();       // my tile-t DMA complete
        __syncthreads();     // everyone's DMA complete; prev buffer free
        if (t0 + 32 < SEQ)   // overlap next tile's DMA with this tile's WMMAs
            stage_kv_async(t0 + 32, bufi ^ 1);

        const __bf16* ksb = ks[bufi];
        const __bf16* vsb = vs[bufi];

        // ---- scores: two 16x16 tiles (kv cols t0..t0+15 and +16..+31) ----
        f32x8 sa = {}, sb = {};
        const char* ab  = (const char*)qs + (wave * 16 + l16) * (LD264 * 2);
        const char* kba = (const char*)ksb + l16 * (LD264 * 2);
        const char* kbb = (const char*)ksb + (16 + l16) * (LD264 * 2);
#pragma unroll
        for (int kt = 0; kt < 8; ++kt) {
            FragBF af, ba, bb2;
            af.u[0]  = *(const uint4*)(ab + kt * 64 + 16 * half);
            af.u[1]  = *(const uint4*)(ab + kt * 64 + 32 + 16 * half);
            ba.u[0]  = *(const uint4*)(kba + kt * 64 + 16 * half);
            ba.u[1]  = *(const uint4*)(kba + kt * 64 + 32 + 16 * half);
            bb2.u[0] = *(const uint4*)(kbb + kt * 64 + 16 * half);
            bb2.u[1] = *(const uint4*)(kbb + kt * 64 + 32 + 16 * half);
            sa = WMMA_BF16(af.v, ba.v, sa);
            sb = WMMA_BF16(af.v, bb2.v, sb);
        }

        // ---- online softmax (per row r; 16-lane groups share a row) ----
        float scl[8];
#pragma unroll
        for (int r = 0; r < 8; ++r) {
            float mx = fmaxf(sa[r], sb[r]);
#pragma unroll
            for (int off = 8; off >= 1; off >>= 1)
                mx = fmaxf(mx, __shfl_xor(mx, off, 32));
            float mn = fmaxf(m_run[r], mx);
            scl[r]   = __builtin_amdgcn_exp2f((m_run[r] - mn) * LOG2E);
            float pa = __builtin_amdgcn_exp2f((sa[r] - mn) * LOG2E);
            float pb = __builtin_amdgcn_exp2f((sb[r] - mn) * LOG2E);
            float rs = pa + pb;
#pragma unroll
            for (int off = 8; off >= 1; off >>= 1)
                rs += __shfl_xor(rs, off, 32);
            l_run[r] = l_run[r] * scl[r] + rs;
            m_run[r] = mn;
            sa[r] = pa; sb[r] = pb;
        }
#pragma unroll
        for (int et = 0; et < 16; ++et)
#pragma unroll
            for (int r = 0; r < 8; ++r) acc[et][r] *= scl[r];

        // ---- transpose P (C layout -> A layout) via wave-private LDS ----
        {
            __bf16* pw = ps + wave * 16 * LD40;
#pragma unroll
            for (int r = 0; r < 8; ++r) {
                int row = r + 8 * half;
                pw[row * LD40 + l16]      = (__bf16)sa[r];
                pw[row * LD40 + 16 + l16] = (__bf16)sb[r];
            }
        }
        FragBF pf;
        {
            const char* pb_ = (const char*)ps + wave * 16 * (LD40 * 2) + l16 * (LD40 * 2);
            pf.u[0] = *(const uint4*)(pb_ + 16 * half);
            pf.u[1] = *(const uint4*)(pb_ + 32 + 16 * half);
        }

        // ---- ctx += P (16x32) @ V (32x256) ----
#pragma unroll
        for (int et = 0; et < 16; ++et) {
            FragBF vf;
            const char* vb = (const char*)vsb + (et * 16 + l16) * (LD40 * 2);
            vf.u[0] = *(const uint4*)(vb + 16 * half);
            vf.u[1] = *(const uint4*)(vb + 32 + 16 * half);
            acc[et] = WMMA_BF16(pf.v, vf.v, acc[et]);
        }
    }

    // ---- epilogue: normalize and store f32 ----
    float linv[8];
#pragma unroll
    for (int r = 0; r < 8; ++r) linv[r] = __builtin_amdgcn_rcpf(l_run[r]);
#pragma unroll
    for (int et = 0; et < 16; ++et) {
#pragma unroll
        for (int r = 0; r < 8; ++r) {
            int s = s0 + wave * 16 + r + 8 * half;
            int e = et * 16 + l16;
            out[((size_t)b * SEQ + s) * EMB + e] = acc[et][r] * linv[r];
        }
    }
}

// ---------------------------------------------------------------------------
extern "C" void kernel_launch(void* const* d_in, const int* in_sizes, int n_in,
                              void* d_out, int out_size, void* d_ws, size_t ws_size,
                              hipStream_t stream) {
    (void)in_sizes; (void)n_in; (void)out_size; (void)ws_size;
    const float* x  = (const float*)d_in[0];
    const float* Wq = (const float*)d_in[1];
    const float* bq = (const float*)d_in[2];
    const float* Wk = (const float*)d_in[3];
    const float* bk = (const float*)d_in[4];
    const float* Wv = (const float*)d_in[5];
    const float* bv = (const float*)d_in[6];
    float* out = (float*)d_out;

    const size_t QK_ELEMS = (size_t)BATCH * SEQ * EMB;      // 16.7M each
    __bf16* qws  = (__bf16*)d_ws;
    __bf16* kws  = qws + QK_ELEMS;
    __bf16* vtws = kws + QK_ELEMS;                          // ~100 MB total

    // Stage 1: QKV projections (bf16 intermediates, V transposed)
    qkv_proj_kernel<<<dim3(BATCH * SEQ / 128, 12), 256, 0, stream>>>(
        x, Wq, bq, Wk, bk, Wv, bv, qws, kws, vtws);

    // Stage 2: fused flash attention
    flash_attn_kernel<<<dim3(SEQ / 128, BATCH), 256, 0, stream>>>(
        qws, kws, vtws, out);
}
